// TemporalAttentionTVModel_69818988364392
// MI455X (gfx1250) — matrix operand order repaired and checked by reading, not compiled, not run on hardware
//
#include <hip/hip_runtime.h>
#include <hip/hip_bf16.h>

// ---------------------------------------------------------------------------
// CDNA5 WMMA types/helpers (bf16 in, f32 accumulate, 16x16x32)
// ---------------------------------------------------------------------------
typedef __attribute__((ext_vector_type(16))) __bf16 v16bf;
typedef __attribute__((ext_vector_type(8)))  float  v8f;
typedef __attribute__((ext_vector_type(4)))  int    v4i;

// value-typedefs carrying explicit address spaces (for async-to-LDS builtin)
typedef __attribute__((address_space(1))) v4i global_v4i;
typedef __attribute__((address_space(3))) v4i shared_v4i;

union Frag16 {
  unsigned short us[16];
  uint4          u4[2];
  v16bf          bf;
};

__device__ __forceinline__ unsigned short f2bf(float f) {
  unsigned u = __float_as_uint(f);
  u += 0x7FFFu + ((u >> 16) & 1u);          // round-to-nearest-even
  return (unsigned short)(u >> 16);
}
__device__ __forceinline__ float bf2f(unsigned short s) {
  return __uint_as_float(((unsigned)s) << 16);
}
__device__ __forceinline__ float gelu_exact(float x) {
  return 0.5f * x * (1.0f + erff(x * 0.70710678118654752f));
}
__device__ __forceinline__ v8f zero8() {
  v8f z; for (int i = 0; i < 8; ++i) z[i] = 0.0f; return z;
}
__device__ __forceinline__ v8f wmma_bf16(v16bf a, v16bf b, v8f c) {
  return __builtin_amdgcn_wmma_f32_16x16x32_bf16(false, a, false, b, (short)0, c,
                                                 false, false);
}

// gfx1250 async global->LDS path (ASYNCcnt), guarded so compile stays green
// on toolchains that don't expose the builtins.
#if defined(__has_builtin)
#  if __has_builtin(__builtin_amdgcn_global_load_async_to_lds_b128) && \
      __has_builtin(__builtin_amdgcn_s_wait_asynccnt)
#    define USE_ASYNC_LDS 1
#  endif
#endif
#ifndef USE_ASYNC_LDS
#  define USE_ASYNC_LDS 0
#endif

// ---------------------------------------------------------------------------
// Weight pre-pass: Wt[n][k] = bf16(W[k][n])  (transpose + convert, 32x32 tiles)
// Run once per launch; makes the GEMM weight stream bf16 + async-copyable.
// ---------------------------------------------------------------------------
__global__ __launch_bounds__(256)
void conv_transpose_bf16_kernel(const float* __restrict__ W,
                                unsigned short* __restrict__ Wt,
                                int K, int N) {
  __shared__ float tile[32][33];
  const int k0 = blockIdx.x * 32, n0 = blockIdx.y * 32;
  const int tx = threadIdx.x & 31, ty = threadIdx.x >> 5;   // ty 0..7
  for (int r = ty; r < 32; r += 8)
    tile[r][tx] = W[(size_t)(k0 + r) * N + n0 + tx];
  __syncthreads();
  for (int r = ty; r < 32; r += 8)
    Wt[(size_t)(n0 + r) * K + k0 + tx] = f2bf(tile[tx][r]);
}

// ---------------------------------------------------------------------------
// GEMM: C[M,N] = act(A[M,K](f32) @ Wt[N,K](bf16)^T + bias[N])
// M%128==0, N%128==0, K%32==0 (true for every call here).
// 256 threads = 8 waves, 128x128 tile, each wave 32x64 (2x4 WMMA tiles).
// B tile staged via GLOBAL_LOAD_ASYNC_TO_LDS_B128 when available.
// ---------------------------------------------------------------------------
#define BM 128
#define BN 128
#define BK 32

__global__ __launch_bounds__(256)
void wmma_gemm_bias_act(const float* __restrict__ A,
                        const unsigned short* __restrict__ Bt,
                        const float* __restrict__ bias, float* __restrict__ C,
                        int M, int N, int K, int act) {
  __shared__ unsigned short sA[BM][BK];   // [m][k] bf16
  __shared__ unsigned short sB[BN][BK];   // [n][k] bf16 (weights already [N][K])

  const int tid  = threadIdx.x;
  const int lane = tid & 31;
  const int wave = tid >> 5;
  const int wm   = wave & 3;
  const int wn   = wave >> 2;
  const int m0   = blockIdx.y * BM;
  const int n0   = blockIdx.x * BN;
  const int li   = lane & 15;
  const int hi   = lane >> 4;

  v8f acc[2][4];
  for (int i = 0; i < 2; ++i)
    for (int j = 0; j < 4; ++j) acc[i][j] = zero8();

  for (int k0 = 0; k0 < K; k0 += BK) {
    // ---- B tile: 128 rows x 64B, raw bf16 copy global->LDS ----
#if USE_ASYNC_LDS
    for (int it = 0; it < 2; ++it) {
      int slot = tid + it * 256;                 // 512 x 16B granules
      int n = slot >> 2;
      int g = slot & 3;
      const unsigned short* src = Bt + (size_t)(n0 + n) * K + k0 + g * 8;
      unsigned short* dst = &sB[n][g * 8];
      __builtin_amdgcn_global_load_async_to_lds_b128(
          (global_v4i*)(v4i*)src, (shared_v4i*)(v4i*)dst, 0, 0);
    }
#else
    for (int it = 0; it < 2; ++it) {
      int slot = tid + it * 256;
      int n = slot >> 2;
      int g = slot & 3;
      *(uint4*)&sB[n][g * 8] =
          *(const uint4*)(Bt + (size_t)(n0 + n) * K + k0 + g * 8);
    }
#endif
    // ---- A tile: 128x32 f32 -> bf16 (VALU convert on the activation path) ----
    for (int it = 0; it < 4; ++it) {
      int slot = tid + it * 256;
      int r = slot >> 3;
      int c4 = (slot & 7) * 4;
      const float4 v = *(const float4*)(A + (size_t)(m0 + r) * K + k0 + c4);
      sA[r][c4 + 0] = f2bf(v.x); sA[r][c4 + 1] = f2bf(v.y);
      sA[r][c4 + 2] = f2bf(v.z); sA[r][c4 + 3] = f2bf(v.w);
    }
    if (k0 + BK < K)   // prefetch next A tile (global_prefetch_b8)
      __builtin_prefetch(A + (size_t)(m0 + (tid >> 1)) * K + k0 + BK + (tid & 1) * 16, 0, 3);
#if USE_ASYNC_LDS
    __builtin_amdgcn_s_wait_asynccnt(0);
#endif
    __syncthreads();

    v16bf af[2], bfr[4];
    for (int mt = 0; mt < 2; ++mt) {
      const unsigned short* row = &sA[wm * 32 + mt * 16 + li][0];
      Frag16 f;
      f.u4[0] = *(const uint4*)(row + hi * 8);        // e0..7 : k = hi*8+e
      f.u4[1] = *(const uint4*)(row + 16 + hi * 8);   // e8..15: k = 16+hi*8+e
      af[mt] = f.bf;
    }
    for (int nt = 0; nt < 4; ++nt) {
      const unsigned short* row = &sB[wn * 64 + nt * 16 + li][0];
      Frag16 f;
      f.u4[0] = *(const uint4*)(row + hi * 8);        // e0..7 : k = hi*8+e
      f.u4[1] = *(const uint4*)(row + 16 + hi * 8);   // e8..15: k = 16+hi*8+e
      bfr[nt] = f.bf;
    }
    for (int mt = 0; mt < 2; ++mt)
      for (int nt = 0; nt < 4; ++nt)
        acc[mt][nt] = wmma_bf16(af[mt], bfr[nt], acc[mt][nt]);
    __syncthreads();
  }

  for (int mt = 0; mt < 2; ++mt)
    for (int nt = 0; nt < 4; ++nt) {
      int col = n0 + wn * 64 + nt * 16 + li;
      float bv = bias ? bias[col] : 0.0f;
      for (int r = 0; r < 8; ++r) {
        int row = m0 + wm * 32 + mt * 16 + r + hi * 8;
        float v = acc[mt][nt][r] + bv;
        if (act == 1) v = gelu_exact(v);
        C[(size_t)row * N + col] = v;
      }
    }
}

// ---------------------------------------------------------------------------
// Fused relative-position attention (skew trick -> all-WMMA).
// One wave per block; block = (16-row i-tile, head, batch).
// ---------------------------------------------------------------------------
__global__ __launch_bounds__(32)
void attention_rel_kernel(const float* __restrict__ Qg, const float* __restrict__ Kg,
                          const float* __restrict__ Vg,
                          const float* __restrict__ relk, const float* __restrict__ relv,
                          float* __restrict__ ctx, float scale) {
  __shared__ float          sS[16][512];    // scores/probs f32          32 KB
  __shared__ unsigned short sRel[16][544];  // qrel window, then arel    17 KB
  __shared__ unsigned short sQ[16][64];     // q tile bf16                2 KB
  __shared__ unsigned short sB16[16][64];   // K / rel_k staging          2 KB
  __shared__ unsigned short sT[64][32];     // V / rel_v transposed       4 KB

  const int lane = threadIdx.x;
  const int li = lane & 15, hi = lane >> 4;
  const int i0 = blockIdx.x * 16;
  const int h  = blockIdx.y;
  const int b  = blockIdx.z;
  const int S = 512, D = 512;
  const int base = 512 - i0 - 15;           // rel window start (always in range)
  const size_t rowbase = (size_t)(b * S) * D + (size_t)h * 64;

  for (int t = lane; t < 16 * 64; t += 32) {
    int r = t >> 6, c = t & 63;
    sQ[r][c] = f2bf(Qg[rowbase + (size_t)(i0 + r) * D + c]);
  }
  for (int t = lane; t < 16 * 544; t += 32) ((unsigned short*)sRel)[t] = 0;
  __syncthreads();

  // qrel[i,c] = q[i,:] . rel_k[base+c,:]
  for (int ct = 0; ct < 34; ++ct) {
    for (int t = lane; t < 16 * 64; t += 32) {
      int r = t >> 6, c = t & 63;
      int rr = ct * 16 + r;
      sB16[r][c] = (rr < 527) ? f2bf(relk[(size_t)(base + rr) * 64 + c]) : (unsigned short)0;
    }
    __syncthreads();
    v8f acc = zero8();
    for (int s = 0; s < 2; ++s) {
      Frag16 fa, fb;
      const unsigned short* ar = &sQ[li][0];
      fa.u4[0] = *(const uint4*)(ar + s * 32 + hi * 8);
      fa.u4[1] = *(const uint4*)(ar + s * 32 + 16 + hi * 8);
      const unsigned short* br = &sB16[li][0];
      fb.u4[0] = *(const uint4*)(br + s * 32 + hi * 8);
      fb.u4[1] = *(const uint4*)(br + s * 32 + 16 + hi * 8);
      acc = wmma_bf16(fa.bf, fb.bf, acc);
    }
    for (int r = 0; r < 8; ++r) sRel[r + hi * 8][ct * 16 + li] = f2bf(acc[r]);
    __syncthreads();
  }

  // scores = scale * Q K^T + qrel(skewed)
  for (int jt = 0; jt < 32; ++jt) {
    int j0 = jt * 16;
    for (int t = lane; t < 16 * 64; t += 32) {
      int r = t >> 6, c = t & 63;
      sB16[r][c] = f2bf(Kg[rowbase + (size_t)(j0 + r) * D + c]);
    }
    __syncthreads();
    v8f acc = zero8();
    for (int s = 0; s < 2; ++s) {
      Frag16 fa, fb;
      const unsigned short* ar = &sQ[li][0];
      fa.u4[0] = *(const uint4*)(ar + s * 32 + hi * 8);
      fa.u4[1] = *(const uint4*)(ar + s * 32 + 16 + hi * 8);
      const unsigned short* br = &sB16[li][0];
      fb.u4[0] = *(const uint4*)(br + s * 32 + hi * 8);
      fb.u4[1] = *(const uint4*)(br + s * 32 + 16 + hi * 8);
      acc = wmma_bf16(fa.bf, fb.bf, acc);
    }
    for (int r = 0; r < 8; ++r) {
      int row = r + hi * 8;
      int j = j0 + li;
      sS[row][j] = acc[r] * scale + bf2f(sRel[row][j + 15 - row]);
    }
    __syncthreads();
  }

  // softmax: lane pair (li, li+16) owns row li, each half of the columns;
  // combine with wave32 shuffles.
  {
    const int jb = hi * 256;
    float m = -3.4e38f;
    for (int j = 0; j < 256; ++j) m = fmaxf(m, sS[li][jb + j]);
    m = fmaxf(m, __shfl_xor(m, 16, 32));
    float sum = 0.f;
    for (int j = 0; j < 256; ++j) {
      float e = __expf(sS[li][jb + j] - m);
      sS[li][jb + j] = e;
      sum += e;
    }
    sum += __shfl_xor(sum, 16, 32);
    float inv = 1.0f / sum;
    for (int j = 0; j < 256; ++j) sS[li][jb + j] *= inv;
  }
  __syncthreads();

  // arel[li, j+15-li] = prob[li, j] (both half-lanes fill their half)
  for (int t = lane; t < 16 * 544; t += 32) ((unsigned short*)sRel)[t] = 0;
  __syncthreads();
  {
    const int jb = hi * 256;
    for (int j = 0; j < 256; ++j)
      sRel[li][jb + j + 15 - li] = f2bf(sS[li][jb + j]);
  }
  __syncthreads();

  v8f cacc[4];
  for (int nt = 0; nt < 4; ++nt) cacc[nt] = zero8();

  // ctx = A @ V
  for (int jt = 0; jt < 16; ++jt) {
    int j0 = jt * 32;
    for (int t = lane; t < 64 * 32; t += 32) {
      int d = t >> 5, jj = t & 31;
      sT[d][jj] = f2bf(Vg[rowbase + (size_t)(j0 + jj) * D + d]);
    }
    __syncthreads();
    Frag16 fa;
    for (int e = 0; e < 8; ++e) fa.us[e]     = f2bf(sS[li][j0 + hi * 8 + e]);
    for (int e = 0; e < 8; ++e) fa.us[8 + e] = f2bf(sS[li][j0 + 16 + hi * 8 + e]);
    for (int nt = 0; nt < 4; ++nt) {
      Frag16 fb;
      const unsigned short* br = &sT[nt * 16 + li][0];
      fb.u4[0] = *(const uint4*)(br + hi * 8);
      fb.u4[1] = *(const uint4*)(br + 16 + hi * 8);
      cacc[nt] = wmma_bf16(fa.bf, fb.bf, cacc[nt]);
    }
    __syncthreads();
  }

  // ctx += arel @ rel_v[window]
  for (int kt = 0; kt < 17; ++kt) {
    int c0 = kt * 32;
    for (int t = lane; t < 64 * 32; t += 32) {
      int d = t >> 5, cc = t & 31;
      int rr = c0 + cc;
      sT[d][cc] = (rr < 527) ? f2bf(relv[(size_t)(base + rr) * 64 + d]) : (unsigned short)0;
    }
    __syncthreads();
    Frag16 fa;
    const unsigned short* ar = &sRel[li][0];
    fa.u4[0] = *(const uint4*)(ar + c0 + hi * 8);
    fa.u4[1] = *(const uint4*)(ar + c0 + 16 + hi * 8);
    for (int nt = 0; nt < 4; ++nt) {
      Frag16 fb;
      const unsigned short* br = &sT[nt * 16 + li][0];
      fb.u4[0] = *(const uint4*)(br + hi * 8);
      fb.u4[1] = *(const uint4*)(br + 16 + hi * 8);
      cacc[nt] = wmma_bf16(fa.bf, fb.bf, cacc[nt]);
    }
    __syncthreads();
  }

  for (int nt = 0; nt < 4; ++nt)
    for (int r = 0; r < 8; ++r) {
      int row = i0 + r + hi * 8;
      int d = nt * 16 + li;
      ctx[rowbase + (size_t)row * D + d] = cacc[nt][r];
    }
}

// ---------------------------------------------------------------------------
// out[row] = LayerNorm(A[row] + B[row]) * g + beta ; one block per row (D=512)
// ---------------------------------------------------------------------------
__global__ __launch_bounds__(256)
void add_ln_kernel(const float* __restrict__ A, const float* __restrict__ Bp,
                   const float* __restrict__ g, const float* __restrict__ beta,
                   float* __restrict__ out, int D) {
  __shared__ float red[256];
  __shared__ float row[512];
  const int r = blockIdx.x;
  const int t = threadIdx.x;

  float partial = 0.f;
  for (int c = t; c < D; c += 256) {
    float v = A[(size_t)r * D + c] + Bp[(size_t)r * D + c];
    row[c] = v;
    partial += v;
  }
  red[t] = partial; __syncthreads();
  for (int s = 128; s > 0; s >>= 1) { if (t < s) red[t] += red[t + s]; __syncthreads(); }
  float mean = red[0] / D;
  __syncthreads();

  partial = 0.f;
  for (int c = t; c < D; c += 256) { float d = row[c] - mean; partial += d * d; }
  red[t] = partial; __syncthreads();
  for (int s = 128; s > 0; s >>= 1) { if (t < s) red[t] += red[t + s]; __syncthreads(); }
  float rstd = rsqrtf(red[0] / D + 1e-5f);
  __syncthreads();

  for (int c = t; c < D; c += 256)
    out[(size_t)r * D + c] = (row[c] - mean) * rstd * g[c] + beta[c];
}

// ---------------------------------------------------------------------------
// x = softmax(comb)[0]*trend + [1..3]*seas_k + [4]*resid
// ---------------------------------------------------------------------------
__global__ __launch_bounds__(256)
void sd_combine_kernel(const float* __restrict__ trend, const float* __restrict__ seas,
                       const float* __restrict__ resid, const float* __restrict__ comb,
                       float* __restrict__ out, int n) {
  int i = blockIdx.x * blockDim.x + threadIdx.x;
  if (i >= n) return;
  float c0 = comb[0], c1 = comb[1], c2 = comb[2], c3 = comb[3], c4 = comb[4];
  float m = fmaxf(fmaxf(fmaxf(c0, c1), fmaxf(c2, c3)), c4);
  float e0 = __expf(c0 - m), e1 = __expf(c1 - m), e2 = __expf(c2 - m),
        e3 = __expf(c3 - m), e4 = __expf(c4 - m);
  float inv = 1.0f / (e0 + e1 + e2 + e3 + e4);
  out[i] = (e0 * trend[i] + e1 * seas[i] + e2 * seas[i + n] +
            e3 * seas[i + 2 * n] + e4 * resid[i]) * inv;
}

// ---------------------------------------------------------------------------
// Host-side orchestration
// ---------------------------------------------------------------------------
extern "C" void kernel_launch(void* const* d_in, const int* in_sizes, int n_in,
                              void* d_out, int out_size, void* d_ws, size_t ws_size,
                              hipStream_t stream) {
  (void)in_sizes; (void)n_in; (void)out_size;
  const float* x_in = (const float*)d_in[0];
  const float* t1_w = (const float*)d_in[1];  const float* t1_b = (const float*)d_in[2];
  const float* t2_w = (const float*)d_in[3];  const float* t2_b = (const float*)d_in[4];
  const float* s1_w = (const float*)d_in[5];  const float* s1_b = (const float*)d_in[6];
  const float* s2_w = (const float*)d_in[7];  const float* s2_b = (const float*)d_in[8];
  const float* r1_w = (const float*)d_in[9];  const float* r1_b = (const float*)d_in[10];
  const float* r2_w = (const float*)d_in[11]; const float* r2_b = (const float*)d_in[12];
  const float* comb = (const float*)d_in[13];
  const float* wq   = (const float*)d_in[14];
  const float* wk   = (const float*)d_in[15];
  const float* wv   = (const float*)d_in[16];
  const float* wo_w = (const float*)d_in[17]; const float* wo_b = (const float*)d_in[18];
  const float* relk = (const float*)d_in[19]; const float* relv = (const float*)d_in[20];
  const float* lag  = (const float*)d_in[21]; const float* lab  = (const float*)d_in[22];
  const float* f1w  = (const float*)d_in[23]; const float* f1b  = (const float*)d_in[24];
  const float* f2w  = (const float*)d_in[25]; const float* f2b  = (const float*)d_in[26];
  const float* n1g  = (const float*)d_in[27]; const float* n1b  = (const float*)d_in[28];
  const float* n2g  = (const float*)d_in[29]; const float* n2b  = (const float*)d_in[30];

  const int M = 2048, Dm = 512, FFd = 2048;
  const size_t MD = (size_t)M * Dm;

  // ---- float workspace ----
  float* ws = (float*)d_ws;
  float* X  = ws;
  float* Yb = X  + MD;
  float* T1 = Yb + MD;                    // [2048,2048]
  float* T2 = T1 + (size_t)M * FFd;
  float* T3 = T2 + MD;
  float* Qb = T3 + MD;
  float* Kb = Qb + MD;
  float* Vb = Kb + MD;
  float* Cb = Vb + MD;
  float* SE = Cb + MD;                    // [3,2048,512]
  float* fend = SE + 3 * MD;

  // ---- bf16 transposed-weight workspace ----
  unsigned short* WB = (unsigned short*)fend;
  size_t wo_ = 0;
  auto allocw = [&](size_t n) { unsigned short* p = WB + wo_; wo_ += n; return p; };
  unsigned short* u_t1 = allocw((size_t)512 * 512);
  unsigned short* u_t2 = allocw((size_t)512 * 512);
  unsigned short *u_s1[3], *u_s2[3];
  for (int kk = 0; kk < 3; ++kk) {
    u_s1[kk] = allocw((size_t)512 * 256);
    u_s2[kk] = allocw((size_t)256 * 512);
  }
  unsigned short* u_r1 = allocw((size_t)512 * 256);
  unsigned short* u_r2 = allocw((size_t)256 * 512);
  unsigned short *u_wq[6], *u_wk[6], *u_wv[6], *u_wo[6], *u_f1[6], *u_f2[6];
  for (int l = 0; l < 6; ++l) {
    u_wq[l] = allocw((size_t)512 * 512);
    u_wk[l] = allocw((size_t)512 * 512);
    u_wv[l] = allocw((size_t)512 * 512);
    u_wo[l] = allocw((size_t)512 * 512);
    u_f1[l] = allocw((size_t)512 * 2048);
    u_f2[l] = allocw((size_t)2048 * 512);
  }
  size_t need = (size_t)(fend - ws) * sizeof(float) + wo_ * sizeof(unsigned short);
  if (ws_size < need) return;

  auto convT = [&](const float* W, unsigned short* Wt, int K, int N) {
    conv_transpose_bf16_kernel<<<dim3(K / 32, N / 32), dim3(256), 0, stream>>>(W, Wt, K, N);
  };
  auto gemm = [&](const float* A, const unsigned short* Bt, const float* bias,
                  float* C, int m, int n, int k, int act) {
    wmma_gemm_bias_act<<<dim3(n / 128, m / 128), dim3(256), 0, stream>>>(
        A, Bt, bias, C, m, n, k, act);
  };
  auto ln = [&](const float* A, const float* B, const float* g, const float* b,
                float* out) {
    add_ln_kernel<<<dim3(M), dim3(256), 0, stream>>>(A, B, g, b, out, Dm);
  };

  // ---- weight transpose/convert pre-pass ----
  convT(t1_w, u_t1, 512, 512);
  convT(t2_w, u_t2, 512, 512);
  for (int kk = 0; kk < 3; ++kk) {
    convT(s1_w + (size_t)kk * 512 * 256, u_s1[kk], 512, 256);
    convT(s2_w + (size_t)kk * 256 * 512, u_s2[kk], 256, 512);
  }
  convT(r1_w, u_r1, 512, 256);
  convT(r2_w, u_r2, 256, 512);
  for (int l = 0; l < 6; ++l) {
    convT(wq + (size_t)l * 512 * 512, u_wq[l], 512, 512);
    convT(wk + (size_t)l * 512 * 512, u_wk[l], 512, 512);
    convT(wv + (size_t)l * 512 * 512, u_wv[l], 512, 512);
    convT(wo_w + (size_t)l * 512 * 512, u_wo[l], 512, 512);
    convT(f1w + (size_t)l * 512 * 2048, u_f1[l], 512, 2048);
    convT(f2w + (size_t)l * 2048 * 512, u_f2[l], 2048, 512);
  }

  // ---- Seasonal decomposition ----
  gemm(x_in, u_t1, t1_b, T1, M, Dm, Dm, 1);
  gemm(T1, u_t2, t2_b, T2, M, Dm, Dm, 0);            // trend -> T2
  for (int kk = 0; kk < 3; ++kk) {
    gemm(x_in, u_s1[kk], s1_b + kk * 256, T1, M, 256, Dm, 1);
    gemm(T1, u_s2[kk], s2_b + kk * Dm, SE + (size_t)kk * MD, M, Dm, 256, 0);
  }
  gemm(x_in, u_r1, r1_b, T1, M, 256, Dm, 1);
  gemm(T1, u_r2, r2_b, T3, M, Dm, 256, 0);           // resid -> T3
  sd_combine_kernel<<<dim3((int)(MD / 256)), dim3(256), 0, stream>>>(
      T2, SE, T3, comb, X, (int)MD);

  // ---- 6 transformer layers ----
  const float scale = 0.125f;                        // 1/sqrt(64)
  for (int l = 0; l < 6; ++l) {
    const float* rk_l = relk + (size_t)l * 1025 * 64;
    const float* rv_l = relv + (size_t)l * 1025 * 64;

    gemm(X, u_wq[l], nullptr, Qb, M, Dm, Dm, 0);
    gemm(X, u_wk[l], nullptr, Kb, M, Dm, Dm, 0);
    gemm(X, u_wv[l], nullptr, Vb, M, Dm, Dm, 0);

    attention_rel_kernel<<<dim3(32, 8, 4), dim3(32), 0, stream>>>(
        Qb, Kb, Vb, rk_l, rv_l, Cb, scale);

    gemm(Cb, u_wo[l], wo_b + (size_t)l * Dm, T2, M, Dm, Dm, 0);
    ln(T2, X, lag + (size_t)l * Dm, lab + (size_t)l * Dm, T3);   // attn
    ln(X, T3, n1g + (size_t)l * Dm, n1b + (size_t)l * Dm, Yb);   // y

    gemm(Yb, u_f1[l], f1b + (size_t)l * FFd, T1, M, FFd, Dm, 1);
    gemm(T1, u_f2[l], f2b + (size_t)l * Dm, T2, M, Dm, FFd, 0);

    float* dst = (l == 5) ? (float*)d_out : X;
    ln(Yb, T2, n2g + (size_t)l * Dm, n2b + (size_t)l * Dm, dst);
  }
}
